// _Attention_4638564680377
// MI455X (gfx1250) — matrix operand order repaired and checked by reading, not compiled, not run on hardware
//
#include <hip/hip_runtime.h>
#include <hip/hip_bf16.h>
#include <math.h>

// ---------------------------------------------------------------------------
// MI455X (gfx1250, wave32) fused attention:
//   p_proj = premise @ W1p ; h = tanh(p_proj + (hyp@W1h + b1h + b1p))
//   score  = h @ W2 (masked) ; alpha = softmax(score) ; ctx = alpha @ premise
// Projection GEMM runs on v_wmma_f32_16x16x32_bf16 (bf16 in, fp32 acc).
// tanh uses gfx1250's native V_TANH_F32 trans op (co-executes with WMMA).
// ---------------------------------------------------------------------------

typedef __attribute__((ext_vector_type(16))) __bf16 v16bf;
typedef __attribute__((ext_vector_type(8)))  float  v8f;

#define B_  64
#define S_  2048
#define D_  256
#define H_  256
#define TM  128   // premise rows per block (8 waves x 16 rows)
#define NT  16    // H / 16 column tiles
#define KS  8     // D / 32 k-steps

__device__ __forceinline__ float fast_tanh(float x) {
#if __has_builtin(__builtin_amdgcn_tanhf)
  return __builtin_amdgcn_tanhf(x);        // v_tanh_f32 (gfx1250 native)
#elif __has_builtin(__builtin_amdgcn_tanh_f32)
  return __builtin_amdgcn_tanh_f32(x);
#else
  return tanhf(x);
#endif
}

// ---------------------------------------------------------------------------
// Kernel 0: c[b,h] = hyp[b,:] @ W1h[:,h] + b1h[h] + b1p[h]
// ---------------------------------------------------------------------------
__global__ void __launch_bounds__(H_)
k_hproj(const float* __restrict__ hyp, const float* __restrict__ W1h,
        const float* __restrict__ b1h, const float* __restrict__ b1p,
        float* __restrict__ hvec) {
  const int b = blockIdx.x;
  const int h = threadIdx.x;
  __shared__ float sh[D_];
  sh[h] = hyp[b * D_ + h];
  __syncthreads();
  float acc = b1h[h] + b1p[h];
#pragma unroll 4
  for (int d = 0; d < D_; ++d) acc += sh[d] * W1h[d * H_ + h];
  hvec[b * H_ + h] = acc;
}

// ---------------------------------------------------------------------------
// Kernel 1: fused projection (WMMA bf16) + tanh + score dot + mask.
// Block = 256 threads = 8 wave32s; each wave owns a 16-row m-tile.
// ---------------------------------------------------------------------------
__global__ void __launch_bounds__(256)
k_score(const float* __restrict__ premise, const float* __restrict__ W1p,
        const unsigned char* __restrict__ mask, const float* __restrict__ W2,
        const float* __restrict__ hvec, float* __restrict__ scores) {
  // W1p staged once per block as bf16, pre-swizzled into B-fragment layout:
  // frag(ntile,kstep) -> 32 lanes x 16 contiguous bf16 (one v16bf per lane).
  __shared__ __align__(32) __bf16 ldsW[NT * KS * 512];  // 128 KB
  __shared__ float ldsC[H_];
  __shared__ float ldsW2[H_];

  const int  t        = threadIdx.x;
  const int  blk      = blockIdx.x;
  const int  b        = blk >> 4;              // 16 row-blocks per batch (S/TM)
  const long rowBlock = (long)blk * TM;

  const int  w    = t >> 5;                    // wave id (0..7)
  const int  lane = t & 31;                    // wave32 lane
  const long rowBase = rowBlock + (long)w * 16;
  const long row     = rowBase + (lane & 15);  // A-matrix row for this lane
  const int  klo     = (lane >> 4) << 3;       // A layout: hi lanes start at K+8
  const float* prow  = premise + row * D_ + klo;

  // warm premise lines while W1p staging runs (emits global_prefetch_b8)
  __builtin_prefetch(prow, 0, 0);

  // stage per-batch bias vector and W2
  ldsC[t]  = hvec[b * H_ + t];
  ldsW2[t] = W2[t];

  // stage W1p (fp32 [K=256][N=256]) -> bf16 fragments.
  // B layout (16-bit, 32x16): lanes 0-15 hold K {0..7,16..23}, lanes 16-31 K {8..15,24..31}.
  {
    const float4* src = (const float4*)W1p;
    for (int j = 0; j < 64; ++j) {
      const int f  = t + 256 * j;        // 16384 float4 total
      const int kk = f >> 6;             // K row (64 float4 per row)
      const int n0 = (f & 63) << 2;      // first of 4 consecutive N columns
      const float4 v = src[f];
      const int kl    = kk & 31;
      const int kstep = kk >> 5;
      const int half  = (kl >> 3) & 1;
      const int e     = (kl & 7) + ((kl >> 4) << 3);
      const int ntile = n0 >> 4;         // n0 % 4 == 0 -> same ntile for all 4
      const float vv[4] = {v.x, v.y, v.z, v.w};
#pragma unroll
      for (int i = 0; i < 4; ++i) {
        const int ln = ((n0 + i) & 15) + (half << 4);
        ldsW[(ntile * KS + kstep) * 512 + ln * 16 + e] = (__bf16)vv[i];
      }
    }
  }

  // Build all 8 A-fragments in registers (fp32 global -> bf16).
  // A layout (16-bit, 16x32): lane<16 holds K {0..7,16..23}; lane>=16 {8..15,24..31}.
  v16bf aF[KS];
#pragma unroll
  for (int k = 0; k < KS; ++k) {
    const float4 p0 = *(const float4*)(prow + k * 32);
    const float4 p1 = *(const float4*)(prow + k * 32 + 4);
    const float4 p2 = *(const float4*)(prow + k * 32 + 16);
    const float4 p3 = *(const float4*)(prow + k * 32 + 20);
    v16bf a;
    a[0]  = (__bf16)p0.x; a[1]  = (__bf16)p0.y; a[2]  = (__bf16)p0.z; a[3]  = (__bf16)p0.w;
    a[4]  = (__bf16)p1.x; a[5]  = (__bf16)p1.y; a[6]  = (__bf16)p1.z; a[7]  = (__bf16)p1.w;
    a[8]  = (__bf16)p2.x; a[9]  = (__bf16)p2.y; a[10] = (__bf16)p2.z; a[11] = (__bf16)p2.w;
    a[12] = (__bf16)p3.x; a[13] = (__bf16)p3.y; a[14] = (__bf16)p3.z; a[15] = (__bf16)p3.w;
    aF[k] = a;
  }

  __syncthreads();

  // Per n-tile: 8 WMMAs over K, then fused bias + native-tanh + W2 partial dot.
  float sacc[8];
#pragma unroll
  for (int i = 0; i < 8; ++i) sacc[i] = 0.0f;

  for (int nt = 0; nt < NT; ++nt) {
    v8f acc = {};
    const __bf16* bBase = &ldsW[nt * KS * 512 + lane * 16];
#pragma unroll
    for (int k = 0; k < KS; ++k) {
      const v16bf bF = *(const v16bf*)(bBase + k * 512);
      acc = __builtin_amdgcn_wmma_f32_16x16x32_bf16(
          false, aF[k], false, bF, (short)0, acc, false, false);
    }
    // C/D layout: lane holds one column n, 8 row values (M = i + 8*(lane>=16))
    const float cv  = ldsC[nt * 16 + (lane & 15)];
    const float w2v = ldsW2[nt * 16 + (lane & 15)];
#pragma unroll
    for (int i = 0; i < 8; ++i) {
      const float ht = fast_tanh(acc[i] + cv);   // v_tanh_f32, branch-free
      sacc[i] += ht * w2v;
    }
  }

  // Reduce across the 16 column-lanes of each half-wave, then write scores.
#pragma unroll
  for (int i = 0; i < 8; ++i) {
    float v = sacc[i];
    v += __shfl_xor(v, 1, 16);
    v += __shfl_xor(v, 2, 16);
    v += __shfl_xor(v, 4, 16);
    v += __shfl_xor(v, 8, 16);
    if ((lane & 15) == 0) {
      const long r = rowBase + ((long)(lane >> 4) << 3) + i;  // flat b*S+s
      scores[r] = mask[r] ? -__builtin_inff() : v;
    }
  }
}

// ---------------------------------------------------------------------------
// Kernel 2: row softmax over S=2048 (one block per batch), alpha -> d_out.
// ---------------------------------------------------------------------------
__global__ void __launch_bounds__(256)
k_softmax(const float* __restrict__ scores, float* __restrict__ alpha) {
  __shared__ float red[256];
  const int b = blockIdx.x, t = threadIdx.x;
  const float* rowp = scores + (long)b * S_;
  float v[8];
  float m = -__builtin_inff();
#pragma unroll
  for (int j = 0; j < 8; ++j) { v[j] = rowp[t + 256 * j]; m = fmaxf(m, v[j]); }
  red[t] = m; __syncthreads();
  for (int o = 128; o > 0; o >>= 1) {
    if (t < o) red[t] = fmaxf(red[t], red[t + o]);
    __syncthreads();
  }
  m = red[0]; __syncthreads();
  float s = 0.0f;
#pragma unroll
  for (int j = 0; j < 8; ++j) { v[j] = __expf(v[j] - m); s += v[j]; }
  red[t] = s; __syncthreads();
  for (int o = 128; o > 0; o >>= 1) {
    if (t < o) red[t] += red[t + o];
    __syncthreads();
  }
  const float inv = 1.0f / red[0];
#pragma unroll
  for (int j = 0; j < 8; ++j) alpha[(long)b * S_ + t + 256 * j] = v[j] * inv;
}

// ---------------------------------------------------------------------------
// Kernel 3a: context partials (split-S, deterministic, coalesced along D).
// ---------------------------------------------------------------------------
__global__ void __launch_bounds__(256)
k_ctx_partial(const float* __restrict__ premise, const float* __restrict__ alpha,
              float* __restrict__ part) {
  const int blk = blockIdx.x;            // B*16 blocks
  const int b = blk >> 4, chunk = blk & 15;
  const int t = threadIdx.x;
  __shared__ float sA[128];
  const int s0 = chunk * 128;
  if (t < 128) sA[t] = alpha[(long)b * S_ + s0 + t];
  __syncthreads();
  const float* base = premise + ((long)b * S_ + s0) * D_ + t;
  float acc = 0.0f;
#pragma unroll 4
  for (int s = 0; s < 128; ++s) acc += sA[s] * base[(long)s * D_];
  part[(long)blk * D_ + t] = acc;
}

// Kernel 3b: fold 16 partials -> context in d_out.
__global__ void __launch_bounds__(256)
k_ctx_reduce(const float* __restrict__ part, float* __restrict__ ctx) {
  const int b = blockIdx.x, t = threadIdx.x;
  float acc = 0.0f;
#pragma unroll
  for (int c = 0; c < 16; ++c) acc += part[((long)(b * 16 + c)) * D_ + t];
  ctx[b * D_ + t] = acc;
}

// ---------------------------------------------------------------------------
extern "C" void kernel_launch(void* const* d_in, const int* in_sizes, int n_in,
                              void* d_out, int out_size, void* d_ws, size_t ws_size,
                              hipStream_t stream) {
  (void)in_sizes; (void)n_in; (void)out_size; (void)ws_size;
  const float*         premise = (const float*)d_in[0];   // [B,S,D]
  const float*         hyp     = (const float*)d_in[1];   // [B,D]
  const unsigned char* mask    = (const unsigned char*)d_in[2]; // [B,S] bool
  const float*         W1p     = (const float*)d_in[3];   // [D,H]
  const float*         b1p     = (const float*)d_in[4];   // [H]
  const float*         W1h     = (const float*)d_in[5];   // [D,H]
  const float*         b1h     = (const float*)d_in[6];   // [H]
  const float*         W2      = (const float*)d_in[7];   // [H]

  float* out   = (float*)d_out;
  float* ctx   = out;                 // [B,D]
  float* alpha = out + B_ * D_;       // [B,S]

  float* ws     = (float*)d_ws;
  float* hvec   = ws;                 // B*H
  float* scores = hvec + B_ * H_;     // B*S
  float* part   = scores + B_ * S_;   // B*16*D

  k_hproj      <<<B_,            H_,  0, stream>>>(hyp, W1h, b1h, b1p, hvec);
  k_score      <<<(B_ * S_) / TM, 256, 0, stream>>>(premise, W1p, mask, W2, hvec, scores);
  k_softmax    <<<B_,            256, 0, stream>>>(scores, alpha);
  k_ctx_partial<<<B_ * 16,       256, 0, stream>>>(premise, alpha, part);
  k_ctx_reduce <<<B_,            256, 0, stream>>>(part, ctx);
}
